// RWKV2TimeMix_40346922779183
// MI455X (gfx1250) — matrix operand order repaired and checked by reading, not compiled, not run on hardware
//
#include <hip/hip_runtime.h>

// ---------------- model constants ----------------
#define BATCH 8
#define CTX   2048
#define NEMBD 1024
#define NHEAD 16
#define HSZ   64
#define MROWS (BATCH * CTX)   // 16384 token rows
#define KDIM  1024            // reduction dim of every GEMM
#define NDIM  1024            // output dim of every GEMM

// GEMM tiling: block = 4 waves, each wave computes a 32x64 output patch.
#define BM 128
#define BN 64
#define BK 32
#define KT (KDIM / BK)        // 32 k-steps

typedef __attribute__((ext_vector_type(16))) __bf16   v16bf;
typedef __attribute__((ext_vector_type(8)))  float    v8f;
typedef __attribute__((ext_vector_type(4)))  unsigned v4u;
typedef __attribute__((ext_vector_type(8)))  unsigned v8u;

union Frag {
    v16bf v;
    uint4 q[2];
};

// LDS byte offset of a __shared__ object: generic pointer's low 32 bits are the
// offset within the LDS aperture (ISA 10.2).
__device__ __forceinline__ unsigned ldsoff(const void* p) {
    return (unsigned)(unsigned long long)(uintptr_t)p;
}

// ---------------- Tensor Data Mover descriptors (2D) ----------------
// D# groups per CDNA5 ISA; tracked by TENSORcnt.
// dsz: 0=1B, 1=2B, 2=4B per element.
__device__ __forceinline__ void tdm_build(v4u& g0, v8u& g1,
                                          unsigned lds_addr, const void* gptr,
                                          unsigned tensor_d0, unsigned tensor_d1,
                                          unsigned tile_d0,   unsigned tile_d1,
                                          unsigned stride0_elems, unsigned dsz)
{
    unsigned long long ga = (unsigned long long)(uintptr_t)gptr;
    g0[0] = 1u;                                                // count=1, user descriptor
    g0[1] = lds_addr;                                          // lds_addr [63:32]
    g0[2] = (unsigned)(ga & 0xffffffffu);                      // global_addr[31:0]
    g0[3] = (unsigned)((ga >> 32) & 0x1ffffffu) | (2u << 30);  // addr[56:32] | type=2
    g1[0] = dsz << 16;                                         // data_size, mask=0
    g1[1] = (tensor_d0 & 0xffffu) << 16;                       // tensor_dim0[15:0] @ [63:48]
    g1[2] = (tensor_d0 >> 16) | ((tensor_d1 & 0xffffu) << 16); // dim0 hi | dim1 lo
    g1[3] = (tensor_d1 >> 16) | (tile_d0 << 16);               // dim1 hi | tile_dim0
    g1[4] = tile_d1;                                           // tile_dim1
    g1[5] = stride0_elems;                                     // tensor_dim0_stride[31:0]
    g1[6] = 0u;
    g1[7] = 0u;
}

__device__ __forceinline__ void tdm_load_2d(unsigned lds_addr, const void* gptr,
                                            unsigned td0, unsigned td1,
                                            unsigned t0,  unsigned t1,
                                            unsigned stride0, unsigned dsz)
{
    v4u g0; v8u g1;
    tdm_build(g0, g1, lds_addr, gptr, td0, td1, t0, t1, stride0, dsz);
    v4u z = {0u, 0u, 0u, 0u};
    asm volatile("tensor_load_to_lds %0, %1, %2, %3"
                 :: "s"(g0), "s"(g1), "s"(z), "s"(z) : "memory");
}

__device__ __forceinline__ void tdm_store_2d(unsigned lds_addr, const void* gptr,
                                             unsigned td0, unsigned td1,
                                             unsigned t0,  unsigned t1,
                                             unsigned stride0, unsigned dsz)
{
    v4u g0; v8u g1;
    tdm_build(g0, g1, lds_addr, gptr, td0, td1, t0, t1, stride0, dsz);
    v4u z = {0u, 0u, 0u, 0u};
    asm volatile("tensor_store_from_lds %0, %1, %2, %3"
                 :: "s"(g0), "s"(g1), "s"(z), "s"(z) : "memory");
}

// ---------------- prep: token shift + time_mix, f32 -> bf16 ----------------
__global__ __launch_bounds__(256)
void prep_xm_kernel(const float* __restrict__ x,
                    const float* __restrict__ tm,
                    __bf16* __restrict__ xm)
{
    size_t i = (size_t)blockIdx.x * blockDim.x + threadIdx.x;
    if (i >= (size_t)MROWS * NEMBD) return;
    int    c  = (int)(i & (NEMBD - 1));
    size_t bt = i >> 10;
    int    t  = (int)(bt & (CTX - 1));
    float cur  = x[i];
    float prev = (t > 0) ? x[i - NEMBD] : 0.0f;
    float mix  = tm[c];
    xm[i] = (__bf16)(cur * mix + prev * (1.0f - mix));
}

// ---------------- weight convert f32 -> bf16 ----------------
__global__ __launch_bounds__(256)
void cvt_bf16_kernel(const float* __restrict__ src, __bf16* __restrict__ dst, int n)
{
    int i = blockIdx.x * blockDim.x + threadIdx.x;
    if (i < n) dst[i] = (__bf16)src[i];
}

// ---------------- TDM-staged double-buffered WMMA GEMM ----------------
// out[m,n] = sum_k A[m,k] * W[n,k];  A:[MROWS,KDIM] bf16, W:[NDIM,KDIM] bf16.
// Block = 4 waves. Wave w computes rows [m0+32w, m0+32w+32) x cols [n0, n0+64):
// 2 A fragments x 4 B fragments -> 8 back-to-back wmma per k-step, B reused.
// Wave 0 DMA-stages A (BMxBK) and W (BNxBK) tiles into LDS via tensor_load_to_lds.
// Epilogue: results staged in LDS, written out by one tensor_store_from_lds DMA.
// EP: 0 = exp(min(acc,60)) -> bf16 (k) | 1 = acc*aux -> bf16 (kv, aux TDM-staged) |
//     2 = sigmoid -> bf16 (r)         | 3 = acc -> f32 (output)
template<int EP>
__global__ __launch_bounds__(128)
void gemm_tdm_kernel(const __bf16* __restrict__ A,
                     const __bf16* __restrict__ W,
                     const __bf16* __restrict__ aux,
                     __bf16* __restrict__ outb,
                     float*  __restrict__ outf)
{
    __shared__ __bf16 Asm[2][BM * BK];                          // 2 x 8 KB
    __shared__ __bf16 Wsm[2][BN * BK];                          // 2 x 4 KB
    __shared__ __attribute__((aligned(16))) char CsmRaw[BM * BN * 4];  // 32 KB C staging
    __shared__ __bf16 Auxsm[BM * BN];                           // 16 KB (EP==1)

    const int m0   = blockIdx.x * BM;
    const int n0   = blockIdx.y * BN;
    const int wave = threadIdx.x >> 5;
    const int lane = threadIdx.x & 31;
    const int half = lane >> 4;          // 0 | 1
    const int ncol = lane & 15;
    const int arow = wave * 32 + (lane & 15);

    v8f acc0[4] = {};
    v8f acc1[4] = {};

    if (wave == 0) {
        tdm_load_2d(ldsoff(&Asm[0][0]), A + (size_t)m0 * KDIM, KDIM, MROWS, BK, BM, KDIM, 1);
        tdm_load_2d(ldsoff(&Wsm[0][0]), W + (size_t)n0 * KDIM, KDIM, NDIM, BK, BN, KDIM, 1);
    }

    for (int kt = 0; kt < KT; ++kt) {
        const int cur = kt & 1;
        const int nxt = cur ^ 1;

        if (wave == 0) __builtin_amdgcn_s_wait_tensorcnt(0);  // tile kt landed in LDS
        __syncthreads();                                      // publish; prior reads of nxt done

        if (wave == 0 && kt + 1 < KT) {                       // DMA next tile during compute
            const int k0 = (kt + 1) * BK;
            tdm_load_2d(ldsoff(&Asm[nxt][0]), A + (size_t)m0 * KDIM + k0, KDIM, MROWS, BK, BM, KDIM, 1);
            tdm_load_2d(ldsoff(&Wsm[nxt][0]), W + (size_t)n0 * KDIM + k0, KDIM, NDIM, BK, BN, KDIM, 1);
        }

        // A fragment (16x32): lane<16 -> row=lane, K {0..7,16..23}; lane>=16 -> K {8..15,24..31}
        Frag a0, a1;
        const __bf16* as0 = &Asm[cur][arow * BK + half * 8];
        const __bf16* as1 = as0 + 16 * BK;
        a0.q[0] = *(const uint4*)(as0);
        a0.q[1] = *(const uint4*)(as0 + 16);
        a1.q[0] = *(const uint4*)(as1);
        a1.q[1] = *(const uint4*)(as1 + 16);

        // B fragment (32x16): lane -> col=lane&15, K {0..7,16..23} (half 0) / {8..15,24..31} (half 1)
        Frag b[4];
        #pragma unroll
        for (int j = 0; j < 4; ++j) {
            const __bf16* ws = &Wsm[cur][(j * 16 + ncol) * BK + half * 8];
            b[j].q[0] = *(const uint4*)(ws);
            b[j].q[1] = *(const uint4*)(ws + 16);
        }

        #pragma unroll
        for (int j = 0; j < 4; ++j) {
            acc0[j] = __builtin_amdgcn_wmma_f32_16x16x32_bf16(
                false, a0.v, false, b[j].v, (short)0, acc0[j], false, false);
            acc1[j] = __builtin_amdgcn_wmma_f32_16x16x32_bf16(
                false, a1.v, false, b[j].v, (short)0, acc1[j], false, false);
        }
    }

    // ---- epilogue: stage activated tile in LDS, DMA out with TDM store ----
    if (EP == 1) {                                            // stage matching k tile
        if (wave == 0) {
            tdm_load_2d(ldsoff(&Auxsm[0]), aux + (size_t)m0 * NDIM + n0,
                        NDIM, MROWS, BN, BM, NDIM, 1);
            __builtin_amdgcn_s_wait_tensorcnt(0);
        }
        __syncthreads();
    }

    float*  Cf = (float*)CsmRaw;
    __bf16* Cb = (__bf16*)CsmRaw;

    // C/D layout: lane holds (M = r + 8*half, N = ncol) in acc element r.
    #pragma unroll
    for (int s = 0; s < 2; ++s) {
        #pragma unroll
        for (int j = 0; j < 4; ++j) {
            #pragma unroll
            for (int r = 0; r < 8; ++r) {
                int ml = wave * 32 + s * 16 + r + half * 8;
                int nl = j * 16 + ncol;
                float val = (s == 0) ? acc0[j][r] : acc1[j][r];
                if (EP == 0) {
                    Cb[ml * BN + nl] = (__bf16)__expf(fminf(val, 60.0f));
                } else if (EP == 1) {
                    Cb[ml * BN + nl] = (__bf16)(val * (float)Auxsm[ml * BN + nl]);
                } else if (EP == 2) {
                    Cb[ml * BN + nl] = (__bf16)(1.0f / (1.0f + __expf(-val)));
                } else {
                    Cf[ml * BN + nl] = val;
                }
            }
        }
    }

    __syncthreads();                                          // tile complete in LDS
    if (wave == 0) {
        if (EP == 3)
            tdm_store_2d(ldsoff(Cf), outf + (size_t)m0 * NDIM + n0,
                         NDIM, MROWS, BN, BM, NDIM, 2);
        else
            tdm_store_2d(ldsoff(Cb), outb + (size_t)m0 * NDIM + n0,
                         NDIM, MROWS, BN, BM, NDIM, 1);
    }
    // s_endpgm performs an implicit wait-idle (drains TENSORcnt); LDS persists
    // until the whole workgroup retires, so the DMA source stays valid.
}

// ---------------- WKV linear-recurrence scan ----------------
// wkv[t]  = tf*kv[t] + Akv ;  Akv <- kv[t] + w*Akv   (w = exp(-exp(time_decay)))
// sumk[t] = tf*k[t]  + Ak  ;  Ak  <- k[t]  + w*Ak
// rwkv = sigmoid(r) * wkv / (sumk + eps), written in place over the r buffer.
__global__ __launch_bounds__(256)
void wkv_scan_kernel(const __bf16* __restrict__ kbuf,
                     const __bf16* __restrict__ kvbuf,
                     __bf16* __restrict__ rbuf,
                     const float* __restrict__ time_decay,
                     const float* __restrict__ time_first)
{
    int tid = blockIdx.x * blockDim.x + threadIdx.x;   // 0 .. B*NDIM-1
    int b  = tid >> 10;
    int hd = tid & 1023;
    int h  = hd >> 6;

    float w  = __expf(-__expf(time_decay[h]));
    float tf = __expf(time_first[h]);

    float Akv = 0.0f, Ak = 0.0f;
    size_t base = (size_t)b * CTX * NDIM + hd;

    for (int t = 0; t < CTX; ++t) {
        size_t idx = base + (size_t)t * NDIM;
        float kv = (float)kvbuf[idx];
        float kk = (float)kbuf[idx];
        float r  = (float)rbuf[idx];
        float wkv = tf * kv + Akv;
        float sk  = tf * kk + Ak + 1e-16f;
        rbuf[idx] = (__bf16)(r * wkv / sk);
        Akv = kv + w * Akv;
        Ak  = kk + w * Ak;
    }
}

// ---------------- launch ----------------
extern "C" void kernel_launch(void* const* d_in, const int* in_sizes, int n_in,
                              void* d_out, int out_size, void* d_ws, size_t ws_size,
                              hipStream_t stream)
{
    const float* x   = (const float*)d_in[0];
    const float* td  = (const float*)d_in[1];
    const float* tfi = (const float*)d_in[2];
    const float* tm  = (const float*)d_in[3];
    const float* Wk  = (const float*)d_in[4];
    const float* Wv  = (const float*)d_in[5];
    const float* Wr  = (const float*)d_in[6];
    const float* Wo  = (const float*)d_in[7];
    float* out = (float*)d_out;

    char* ws = (char*)d_ws;
    size_t off = 0;
    const size_t MAT = (size_t)MROWS * NDIM;       // 16M elements
    const size_t WSZ = (size_t)NDIM * KDIM;        // 1M elements

    __bf16* xm   = (__bf16*)(ws + off); off += MAT * 2;   // 32 MB
    __bf16* WkB  = (__bf16*)(ws + off); off += WSZ * 2;   //  2 MB
    __bf16* WvB  = (__bf16*)(ws + off); off += WSZ * 2;
    __bf16* WrB  = (__bf16*)(ws + off); off += WSZ * 2;
    __bf16* WoB  = (__bf16*)(ws + off); off += WSZ * 2;
    __bf16* kB   = (__bf16*)(ws + off); off += MAT * 2;   // 32 MB
    __bf16* kvB  = (__bf16*)(ws + off); off += MAT * 2;   // 32 MB
    __bf16* rB   = (__bf16*)(ws + off); off += MAT * 2;   // 32 MB (also rwkv)

    // 1) token shift + mix
    prep_xm_kernel<<<dim3((unsigned)((MAT + 255) / 256)), 256, 0, stream>>>(x, tm, xm);

    // 2) weights -> bf16
    {
        dim3 g((unsigned)((WSZ + 255) / 256));
        cvt_bf16_kernel<<<g, 256, 0, stream>>>(Wk, WkB, (int)WSZ);
        cvt_bf16_kernel<<<g, 256, 0, stream>>>(Wv, WvB, (int)WSZ);
        cvt_bf16_kernel<<<g, 256, 0, stream>>>(Wr, WrB, (int)WSZ);
        cvt_bf16_kernel<<<g, 256, 0, stream>>>(Wo, WoB, (int)WSZ);
    }

    // 3) projections (TDM + WMMA GEMMs)
    dim3 gg(MROWS / BM, NDIM / BN);
    gemm_tdm_kernel<0><<<gg, 128, 0, stream>>>(xm, WkB, nullptr, kB, nullptr); // k = exp(min(.,60))
    gemm_tdm_kernel<1><<<gg, 128, 0, stream>>>(xm, WvB, kB, kvB, nullptr);     // kv = k*v
    gemm_tdm_kernel<2><<<gg, 128, 0, stream>>>(xm, WrB, nullptr, rB, nullptr); // sigmoid(r)

    // 4) WKV recurrence scan (rwkv -> rB in place)
    wkv_scan_kernel<<<dim3((BATCH * NDIM) / 256), 256, 0, stream>>>(kB, kvB, rB, td, tfi);

    // 5) output projection -> f32
    gemm_tdm_kernel<3><<<gg, 128, 0, stream>>>(rB, WoB, nullptr, nullptr, out);
}
